// EFNtoGlobal_10943576670837
// MI455X (gfx1250) — matrix operand order repaired and checked by reading, not compiled, not run on hardware
//
#include <hip/hip_runtime.h>
#include <hip/hip_bf16.h>
#include <stdint.h>

// ---------------------------------------------------------------------------
// EFN-to-global on gfx1250: per-edge 2-layer MLP as bf16 WMMA (f32 accum),
// transposed formulation so layer1->layer2 handoff is a lane^16 register
// exchange (no LDS round trip), biases folded into WMMA C-init, LDS padded
// to kill bank conflicts, persistent waves amortize weight staging.
// ---------------------------------------------------------------------------

typedef __attribute__((ext_vector_type(2)))  __bf16 bf16x2;
typedef __attribute__((ext_vector_type(4)))  __bf16 bf16x4;
typedef __attribute__((ext_vector_type(8)))  __bf16 bf16x8;
typedef __attribute__((ext_vector_type(16))) __bf16 v16bf;
typedef __attribute__((ext_vector_type(8)))  float  v8f;
typedef __attribute__((ext_vector_type(8)))  unsigned int v8u;

#define D_FEAT   64
#define HIDDEN   128
#define XP       136          // padded LDS row pitch (272B): +4 banks per row
#define ET       2            // 16-edge sub-tiles per wave iteration
#define TILE_E   (16 * ET)    // 32 edges per wave iteration
#define WPB      8            // waves per block
#define TPB      (32 * WPB)
#define MAX_BLK  1024

__device__ __forceinline__ void wave_lds_fence() {
    asm volatile("s_wait_dscnt 0x0" ::: "memory");
}

__device__ __forceinline__ v16bf cat8(bf16x8 lo, bf16x8 hi) {
    return __builtin_shufflevector(lo, hi, 0, 1, 2, 3, 4, 5, 6, 7,
                                           8, 9, 10, 11, 12, 13, 14, 15);
}

__device__ __forceinline__ v8f wmma_bf16(v16bf a, v16bf b, v8f c) {
    return __builtin_amdgcn_wmma_f32_16x16x32_bf16(false, a, false, b,
                                                   (short)0, c, false, false);
}

__global__ __launch_bounds__(TPB)
void edge_mlp_wmma(const float* __restrict__ x,
                   const float* __restrict__ W1, const float* __restrict__ b1,
                   const float* __restrict__ W2, const float* __restrict__ b2,
                   const int* __restrict__ e_src, const int* __restrict__ e_dst,
                   float* __restrict__ node_out, int n_edges)
{
    // Transposed weights [n][k] with padded pitch XP.
    __shared__ __align__(32) __bf16 sW1T[HIDDEN * XP];        // ~34 KB
    __shared__ __align__(32) __bf16 sW2T[D_FEAT * XP];        // ~17 KB
    __shared__ __align__(32) __bf16 sX[WPB * TILE_E * XP];    // ~68 KB edge tiles
    __shared__ int sDst[WPB * TILE_E];

    const int tid = threadIdx.x;
    // Stage weights: consecutive tids write consecutive k (2B stride, no LDS
    // conflicts); global reads are strided but W1/W2 are L2-resident.
    for (int i = tid; i < HIDDEN * HIDDEN; i += TPB) {
        const int n = i >> 7, k = i & 127;
        sW1T[n * XP + k] = (__bf16)W1[k * HIDDEN + n];
    }
    for (int i = tid; i < D_FEAT * HIDDEN; i += TPB) {
        const int n = i >> 7, k = i & 127;
        sW2T[n * XP + k] = (__bf16)W2[k * D_FEAT + n];
    }
    __syncthreads();

    const int wave = tid >> 5, lane = tid & 31;
    const int nl = lane & 15, half = lane >> 4;
    __bf16* sXw = &sX[wave * TILE_E * XP];
    int*    sDw = &sDst[wave * TILE_E];

    const int tiles  = (n_edges + TILE_E - 1) / TILE_E;
    const int stride = gridDim.x * WPB;

    for (int tw = blockIdx.x * WPB + wave; tw < tiles; tw += stride) {
        const int e0 = tw * TILE_E;

        // ---- Gather 32 edge rows [x[dst] | x[src]] -> bf16 row-major tile ----
        #pragma unroll
        for (int t = 0; t < ET; ++t) {
            const int row = t * 16 + nl;
            int e = e0 + row; if (e >= n_edges) e = n_edges - 1;
            const int node = half ? e_src[e] : e_dst[e];
            if (!half) sDw[row] = node;
            const float4* xr = reinterpret_cast<const float4*>(x + (size_t)node * D_FEAT);
            __bf16* drow = sXw + row * XP + half * D_FEAT;
            #pragma unroll
            for (int j = 0; j < D_FEAT / 4; ++j) {
                const float4 v = xr[j];
                bf16x4 pv = { (__bf16)v.x, (__bf16)v.y, (__bf16)v.z, (__bf16)v.w };
                *reinterpret_cast<bf16x4*>(drow + j * 4) = pv;
            }
        }
        wave_lds_fence();

        // ---- GEMM1 (transposed): preact[n, m] = W1T . xT, C-init = b1 ----
        // A (W1T tile): lane(h,nl) row n=nt*16+nl, K = K0+16h+0..15 (contiguous)
        // B (x tile)  : lane(h,nl) col m=nl,       K = K0+16h+0..15 (contiguous)
        // D           : lane(h,nl) holds preact[n = nt*16 + i + 8h][m = nl]
        v8f acc1[8][ET];
        #pragma unroll
        for (int nt = 0; nt < 8; ++nt) {
            const float4 blo = *reinterpret_cast<const float4*>(b1 + nt * 16 + 8 * half);
            const float4 bhi = *reinterpret_cast<const float4*>(b1 + nt * 16 + 8 * half + 4);
            const v8f c = { blo.x, blo.y, blo.z, blo.w, bhi.x, bhi.y, bhi.z, bhi.w };
            #pragma unroll
            for (int t = 0; t < ET; ++t) acc1[nt][t] = c;
        }
        #pragma unroll
        for (int kk = 0; kk < 4; ++kk) {
            const int K0 = kk * 32;
            v16bf bt[ET];
            #pragma unroll
            for (int t = 0; t < ET; ++t)
                bt[t] = *reinterpret_cast<const v16bf*>(sXw + (t * 16 + nl) * XP + K0 + 16 * half);
            #pragma unroll
            for (int nt = 0; nt < 8; ++nt) {
                const int n = nt * 16 + nl;
                const bf16x8 alo = *reinterpret_cast<const bf16x8*>(&sW1T[n * XP + K0 + 16 * half]);
                const bf16x8 ahi = *reinterpret_cast<const bf16x8*>(&sW1T[n * XP + K0 + 16 * half + 8]);
                const v16bf a = cat8(alo, ahi);
                #pragma unroll
                for (int t = 0; t < ET; ++t)
                    acc1[nt][t] = wmma_bf16(a, bt[t], acc1[nt][t]);
            }
        }

        // ---- ReLU + pack bf16 pairs: pk[nt][j] = H[n=nt*16+8h+2j .. +1][m=nl] ----
        unsigned int pk[8][4][ET];
        #pragma unroll
        for (int nt = 0; nt < 8; ++nt)
            #pragma unroll
            for (int j = 0; j < 4; ++j)
                #pragma unroll
                for (int t = 0; t < ET; ++t) {
                    float lo = acc1[nt][t][2 * j];     lo = lo > 0.f ? lo : 0.f;
                    float hi = acc1[nt][t][2 * j + 1]; hi = hi > 0.f ? hi : 0.f;
                    const bf16x2 pv = { (__bf16)lo, (__bf16)hi };
                    pk[nt][j][t] = __builtin_bit_cast(unsigned int, pv);
                }

        // ---- GEMM2 (transposed): out[n2, m] = W2T . HT, C-init = b2 ----
        v8f acc2[4][ET];
        #pragma unroll
        for (int nt = 0; nt < 4; ++nt) {
            const float4 blo = *reinterpret_cast<const float4*>(b2 + nt * 16 + 8 * half);
            const float4 bhi = *reinterpret_cast<const float4*>(b2 + nt * 16 + 8 * half + 4);
            const v8f c = { blo.x, blo.y, blo.z, blo.w, bhi.x, bhi.y, bhi.z, bhi.w };
            #pragma unroll
            for (int t = 0; t < ET; ++t) acc2[nt][t] = c;
        }
        #pragma unroll
        for (int kk = 0; kk < 4; ++kk) {
            const int K0 = kk * 32;
            // B fragment: lane(h,nl) needs H[32kk+16h + 0..15][m=nl].
            // Low 8 values live on lane(0,nl), high 8 on lane(1,nl), with
            // nt = 2kk + h_dest -> one lane^16 exchange per pair.
            v16bf bfrag[ET];
            #pragma unroll
            for (int t = 0; t < ET; ++t) {
                v8u bu;
                #pragma unroll
                for (int j = 0; j < 4; ++j) {
                    const unsigned a0 = pk[2 * kk][j][t];       // owner h=0 data
                    const unsigned a1 = pk[2 * kk + 1][j][t];   // owner h=1 data
                    const unsigned self = half ? a1 : a0;       // pk[2kk+half]
                    const unsigned sent = half ? a0 : a1;       // pk[2kk+1-half]
                    const unsigned recv =
                        (unsigned)__shfl_xor((int)sent, 16, 32);
                    bu[j]     = half ? recv : self;             // K = 32kk+16h+2j
                    bu[j + 4] = half ? self : recv;             // K = +8..15
                }
                bfrag[t] = __builtin_bit_cast(v16bf, bu);
            }
            #pragma unroll
            for (int nt = 0; nt < 4; ++nt) {
                const int n2 = nt * 16 + nl;
                const bf16x8 alo = *reinterpret_cast<const bf16x8*>(&sW2T[n2 * XP + K0 + 16 * half]);
                const bf16x8 ahi = *reinterpret_cast<const bf16x8*>(&sW2T[n2 * XP + K0 + 16 * half + 8]);
                const v16bf a = cat8(alo, ahi);
                #pragma unroll
                for (int t = 0; t < ET; ++t)
                    acc2[nt][t] = wmma_bf16(a, bfrag[t], acc2[nt][t]);
            }
        }

        // ---- Scatter: lane(h,nl) owns out[f = nt*16+i+8h][edge = t*16+nl] ----
        #pragma unroll
        for (int t = 0; t < ET; ++t) {
            const int row = t * 16 + nl;
            if (e0 + row < n_edges) {
                const int node = sDw[row];
                float* np = node_out + (size_t)node * D_FEAT + 8 * half;
                #pragma unroll
                for (int nt = 0; nt < 4; ++nt)
                    #pragma unroll
                    for (int i = 0; i < 8; ++i)
                        atomicAdd(np + nt * 16 + i, acc2[nt][t][i]);
            }
        }
    }
}

// weighted = p[:,0] * node_out ; out[g] += weighted (global_add_pool)
__global__ __launch_bounds__(256)
void pool_kernel(const float* __restrict__ node_out, const float* __restrict__ p,
                 const int* __restrict__ batch, float* __restrict__ out, int n_nodes)
{
    const int idx  = blockIdx.x * blockDim.x + threadIdx.x;
    const int node = idx >> 6;
    const int f    = idx & 63;
    if (node < n_nodes) {
        const float v = p[(size_t)node * 4] * node_out[(size_t)node * D_FEAT + f];
        atomicAdd(out + (size_t)batch[node] * D_FEAT + f, v);
    }
}

extern "C" void kernel_launch(void* const* d_in, const int* in_sizes, int n_in,
                              void* d_out, int out_size, void* d_ws, size_t ws_size,
                              hipStream_t stream)
{
    const float* x    = (const float*)d_in[0];
    const float* p    = (const float*)d_in[1];
    const float* W1   = (const float*)d_in[2];
    const float* b1   = (const float*)d_in[3];
    const float* W2   = (const float*)d_in[4];
    const float* b2   = (const float*)d_in[5];
    const int*   eidx = (const int*)d_in[6];   // [2, E] flat: src row then dst row
    const int*   batch= (const int*)d_in[7];
    float* out = (float*)d_out;

    const int n_edges = in_sizes[6] / 2;
    const int n_nodes = in_sizes[0] / D_FEAT;

    float* node_out = (float*)d_ws;            // [n_nodes, 64] f32 scratch

    hipMemsetAsync(node_out, 0, (size_t)n_nodes * D_FEAT * sizeof(float), stream);
    hipMemsetAsync(d_out, 0, (size_t)out_size * sizeof(float), stream);

    const int tiles = (n_edges + TILE_E - 1) / TILE_E;
    int blocks = (tiles + WPB - 1) / WPB;
    if (blocks > MAX_BLK) blocks = MAX_BLK;

    edge_mlp_wmma<<<blocks, TPB, 0, stream>>>(
        x, W1, b1, W2, b2, eidx, eidx + n_edges, node_out, n_edges);

    const int total = n_nodes * D_FEAT;
    pool_kernel<<<(total + 255) / 256, 256, 0, stream>>>(node_out, p, batch, out, n_nodes);
}
